// WaveGlow_49641232007718
// MI455X (gfx1250) — matrix operand order repaired and checked by reading, not compiled, not run on hardware
//
#include <hip/hip_runtime.h>
#include <hip/hip_bf16.h>

typedef __attribute__((ext_vector_type(16))) _Float16 v16h;
typedef __attribute__((ext_vector_type(8)))  _Float16 v8h;
typedef __attribute__((ext_vector_type(8)))  float    v8f;
typedef __attribute__((ext_vector_type(4)))  float    v4f;

#define NB   2
#define NMEL 80
#define TT   32768
#define LG   4096
#define NPOS 8192      /* NB*LG */
#define NCH  128

// ---------------------------------------------------------------------------
// WMMA GEMM core:  acc[j] += A-tile x B-tile over K = TAPS*CIN
// A: f16 [M][K] tap-major.  Bact: f16 position-major [NPOS][CIN].
// CHECK=false: caller guarantees every tap-shifted row is in [0,LG).
// ---------------------------------------------------------------------------
template <int TAPS, int CIN, bool CHECK>
__device__ __forceinline__ void gemm_core(
    const _Float16* __restrict__ A, const _Float16* __restrict__ Bact,
    int m0, int n0, int dil, v8f acc[4], int lm, int half)
{
  constexpr int K   = TAPS * CIN;
  constexpr int NC  = CIN / 32;
  constexpr int PAD = TAPS / 2;
  const int bb = n0 >> 12;                  // batch (L = 4096)
  const int l0 = n0 & (LG - 1);
  const _Float16* arow = A + (size_t)(m0 + lm) * K + half * 8;

#pragma unroll
  for (int t = 0; t < TAPS; ++t) {
    const int dl = (t - PAD) * dil;
    const _Float16* bp[4];
    bool ok[4];
#pragma unroll
    for (int j = 0; j < 4; ++j) {
      const int l = l0 + j * 16 + lm + dl;
      ok[j] = !CHECK || ((unsigned)l < (unsigned)LG);
      bp[j] = Bact + (size_t)((bb << 12) + l) * CIN + half * 8;
    }
    const _Float16* ap = arow + t * CIN;

    auto chunk = [&]() {
      // A fragment: 16x32 f16, lane m=m0+lm, K-within-chunk {half*8+0..7, 16+half*8+0..7}
      v8h alo = *(const v8h*)(ap);
      v8h ahi = *(const v8h*)(ap + 16);
      ap += 32;
      v16h af = __builtin_shufflevector(alo, ahi,
          0,1,2,3,4,5,6,7,8,9,10,11,12,13,14,15);
#pragma unroll
      for (int j = 0; j < 4; ++j) {
        // B fragment: 32x16 f16, lane col n, K-within-chunk {half*8+0..7, 16+half*8+0..7}
        v16h bf = {};
        if (ok[j]) {
          v8h b0 = *(const v8h*)(bp[j]);
          v8h b1 = *(const v8h*)(bp[j] + 16);
          bf = __builtin_shufflevector(b0, b1,
              0,1,2,3,4,5,6,7,8,9,10,11,12,13,14,15);
        }
        bp[j] += 32;
        acc[j] = __builtin_amdgcn_wmma_f32_16x16x32_f16(
            false, af, false, bf, (short)0, acc[j], false, false);
      }
    };

    if constexpr (NC <= 4) {
#pragma unroll
      for (int cc = 0; cc < NC; ++cc) chunk();
    } else {
#pragma unroll 2
      for (int cc = 0; cc < NC; ++cc) chunk();
    }
  }
}

// out[n][m] = bias[m] + (optional add[n][addOff+m]) + conv-GEMM
// Block = 128 threads (4 waves). Wave tile = 16(M) x 64(N). Grid = (M/64, NPOS/64).
template <int TAPS, int CIN>
__global__ __launch_bounds__(128) void gemm_wmma(
    const _Float16* __restrict__ A, const _Float16* __restrict__ Bact,
    const float* __restrict__ bias, const float* __restrict__ addsrc,
    float* __restrict__ out, int dil, int addStride, int addOff, int outStride)
{
  const int lane = threadIdx.x & 31;
  const int wave = threadIdx.x >> 5;
  const int lm   = lane & 15;
  const int half = lane >> 4;
  const int m0   = blockIdx.x * 64 + wave * 16;
  const int n0   = blockIdx.y * 64;
  const int l0   = n0 & (LG - 1);

  v8f acc[4] = {};
  const int margin = (TAPS / 2) * dil;
  if (l0 >= margin && l0 + 63 + margin < LG)     // block-uniform branch
    gemm_core<TAPS, CIN, false>(A, Bact, m0, n0, dil, acc, lm, half);
  else
    gemm_core<TAPS, CIN, true >(A, Bact, m0, n0, dil, acc, lm, half);

  // Epilogue: acc element v -> row m = m0 + half*8 + v, col n = n0 + j*16 + lm
  const int mb = m0 + half * 8;
  v4f bia0 = *(const v4f*)(bias + mb);
  v4f bia1 = *(const v4f*)(bias + mb + 4);
#pragma unroll
  for (int j = 0; j < 4; ++j) {
    const int n = n0 + j * 16 + lm;
    v4f r0 = __builtin_shufflevector(acc[j], acc[j], 0, 1, 2, 3) + bia0;
    v4f r1 = __builtin_shufflevector(acc[j], acc[j], 4, 5, 6, 7) + bia1;
    if (addsrc) {
      const float* ap = addsrc + (size_t)n * addStride + addOff + mb;
      r0 += *(const v4f*)ap;
      r1 += *(const v4f*)(ap + 4);
    }
    float* op = out + (size_t)n * outStride + mb;
    *(v4f*)op       = r0;
    *(v4f*)(op + 4) = r1;
  }
}

// ---------------------------------------------------------------------------
// Transposed-conv upsample: su[b][o][t] = upb[o] + sum_{i,f} spect[b][i][f]*w[i][o][t-256f]
// ---------------------------------------------------------------------------
__global__ void upsample_kernel(const float* __restrict__ spect,
                                const float* __restrict__ w,
                                const float* __restrict__ b,
                                float* __restrict__ su)
{
  int idx = blockIdx.x * blockDim.x + threadIdx.x;
  if (idx >= NB * NMEL * TT) return;
  int t = idx % TT, rem = idx / TT;
  int o = rem % NMEL, bb = rem / NMEL;
  float acc = b[o];
  int f_hi = t >> 8; if (f_hi > 127) f_hi = 127;
  int f_lo = (t - 768) >> 8; if (f_lo < 0) f_lo = 0;   // ceil((t-1023)/256)
  for (int f = f_lo; f <= f_hi; ++f) {
    int k = t - (f << 8);
    const float* sp = spect + (size_t)bb * NMEL * 128 + f;
    const float* wp = w + (size_t)o * 1024 + k;
    for (int i = 0; i < NMEL; ++i)
      acc += sp[i * 128] * wp[(size_t)i * NMEL * 1024];
  }
  su[idx] = acc;
}

// spect_g f16, position-major: sgh[(b*L+l)*640 + m*8+g] = su[b][m][l*8+g]
__global__ void pack_sg_kernel(const float* __restrict__ su, _Float16* __restrict__ sgh)
{
  int idx = blockIdx.x * blockDim.x + threadIdx.x;
  if (idx >= NPOS * 640) return;
  int ci = idx % 640, n = idx / 640;
  int l = n & (LG - 1), bb = n >> 12;
  int m = ci >> 3, g = ci & 7;
  sgh[idx] = (_Float16)su[(((size_t)bb * NMEL + m) << 15) + (l << 3) + g];
}

// a[b][g][l] = audio[b][l*8+g]  (channel-major, 8 channels packed)
__global__ void pack_audio_kernel(const float* __restrict__ audio, float* __restrict__ a)
{
  int idx = blockIdx.x * blockDim.x + threadIdx.x;
  if (idx >= NB * 8 * LG) return;
  int l = idx & (LG - 1), rem = idx >> 12;
  int g = rem & 7, bb = rem >> 3;
  a[idx] = audio[((size_t)bb << 15) + (l << 3) + g];
}

// Repack conv weight [Cout][Cin][taps] (fp32) -> [Cout][t*Cin+ci] (f16)
__global__ void cvtw_kernel(const float* __restrict__ src, _Float16* __restrict__ dst,
                            int Cin, int taps, int total)
{
  int idx = blockIdx.x * blockDim.x + threadIdx.x;
  if (idx >= total) return;
  int ktot = Cin * taps;
  int co = idx / ktot, r = idx % ktot;
  int t = r / Cin, ci = r % Cin;
  dst[idx] = (_Float16)src[(size_t)(co * Cin + ci) * taps + t];
}

// 1-thread log|det| via partial-pivot LU; adds B*L*log|det| into acc.
__global__ void logdet_kernel(const float* __restrict__ W, int n, float* acc)
{
  if (threadIdx.x != 0 || blockIdx.x != 0) return;
  double m[8][8];
  for (int i = 0; i < n; ++i)
    for (int j = 0; j < n; ++j) m[i][j] = (double)W[i * n + j];
  double det = 1.0;
  for (int c = 0; c < n; ++c) {
    int p = c; double best = fabs(m[c][c]);
    for (int r = c + 1; r < n; ++r) { double v = fabs(m[r][c]); if (v > best) { best = v; p = r; } }
    if (p != c) { for (int j = 0; j < n; ++j) { double tm = m[c][j]; m[c][j] = m[p][j]; m[p][j] = tm; } det = -det; }
    double piv = m[c][c]; det *= piv;
    if (piv != 0.0)
      for (int r = c + 1; r < n; ++r) {
        double f = m[r][c] / piv;
        for (int j = c; j < n; ++j) m[r][j] -= f * m[c][j];
      }
  }
  *acc += (float)(8192.0 * log(fabs(det)));
}

// invertible 1x1: anew[b][o][l] = sum_i W[o][i] * a[b][i][l]
__global__ void mix_kernel(const float* __restrict__ W, const float* __restrict__ a,
                           float* __restrict__ o, int nrem)
{
  int idx = blockIdx.x * blockDim.x + threadIdx.x;
  if (idx >= NB * nrem * LG) return;
  int l = idx & (LG - 1), rem = idx >> 12;
  int oc = rem % nrem, bb = rem / nrem;
  float s = 0.f;
  for (int i = 0; i < nrem; ++i)
    s += W[oc * nrem + i] * a[(((size_t)bb * nrem + i) << 12) + l];
  o[idx] = s;
}

// start conv (1x1, nhalf->128) + zero skip; writes x fp32 and f16 copy (position-major)
__global__ void start_kernel(const float* __restrict__ a, const float* __restrict__ sw,
                             const float* __restrict__ sb, float* __restrict__ x,
                             _Float16* __restrict__ xh, float* __restrict__ skip,
                             int nhalf, int nrem)
{
  int idx = blockIdx.x * blockDim.x + threadIdx.x;
  if (idx >= NPOS * NCH) return;
  int c = idx & (NCH - 1), n = idx >> 7;
  int l = n & (LG - 1), bb = n >> 12;
  float s = sb[c];
  for (int i = 0; i < nhalf; ++i)
    s += sw[c * nhalf + i] * a[(((size_t)bb * nrem + i) << 12) + l];
  x[idx] = s; xh[idx] = (_Float16)s; skip[idx] = 0.f;
}

// gated activation: acts = tanh(z[:,0:128]) * sigmoid(z[:,128:256]) -> f16
__global__ void act_kernel(const float* __restrict__ z, _Float16* __restrict__ ah)
{
  int idx = blockIdx.x * blockDim.x + threadIdx.x;
  if (idx >= NPOS * NCH) return;
  int c = idx & (NCH - 1); size_t n = idx >> 7;
  float tz = z[n * 256 + c];
  float sz = z[n * 256 + 128 + c];
  ah[idx] = (_Float16)(tanhf(tz) * (1.f / (1.f + expf(-sz))));
}

// residual + skip update
__global__ void resid_kernel(const float* __restrict__ rs, float* __restrict__ x,
                             _Float16* __restrict__ xh, float* __restrict__ skip, int last)
{
  int idx = blockIdx.x * blockDim.x + threadIdx.x;
  if (idx >= NPOS * NCH) return;
  int c = idx & (NCH - 1); size_t n = idx >> 7;
  if (!last) {
    float nx = x[idx] + rs[n * 256 + c];
    x[idx] = nx; xh[idx] = (_Float16)nx;
    skip[idx] += rs[n * 256 + 128 + c];
  } else {
    skip[idx] += rs[n * 128 + c];
  }
}

// end conv (128 -> 2*nhalf) + affine coupling + log_s reduction
__global__ __launch_bounds__(256) void end_kernel(
    const float* __restrict__ skip, const float* __restrict__ ew,
    const float* __restrict__ eb, float* __restrict__ a,
    int nhalf, int nrem, float* __restrict__ ls_acc)
{
  int n = blockIdx.x * blockDim.x + threadIdx.x;
  float lssum = 0.f;
  if (n < NPOS) {
    int bb = n >> 12, l = n & (LG - 1);
    const float* sp = skip + (size_t)n * NCH;
    for (int j = 0; j < nhalf; ++j) {
      float bt = eb[j], s = eb[nhalf + j];
      for (int ci = 0; ci < NCH; ++ci) {
        float sv = sp[ci];
        bt += ew[j * NCH + ci] * sv;
        s  += ew[(nhalf + j) * NCH + ci] * sv;
      }
      float* ap = a + (((size_t)bb * nrem + nhalf + j) << 12) + l;
      *ap = expf(s) * (*ap) + bt;
      lssum += s;
    }
  }
  __shared__ float red[256];
  red[threadIdx.x] = lssum; __syncthreads();
  for (int off = 128; off > 0; off >>= 1) {
    if ((int)threadIdx.x < off) red[threadIdx.x] += red[threadIdx.x + off];
    __syncthreads();
  }
  if (threadIdx.x == 0) atomicAdd(ls_acc, red[0]);
}

// early split at flow 4: z[:,0:2] = a[:,0:2]; anew (6ch) = a[:,2:8]
__global__ void early_kernel(const float* __restrict__ a, float* __restrict__ z,
                             float* __restrict__ anew)
{
  int idx = blockIdx.x * blockDim.x + threadIdx.x;
  if (idx >= NB * 8 * LG) return;
  int l = idx & (LG - 1), rem = idx >> 12;
  int c = rem & 7, bb = rem >> 3;
  float v = a[idx];
  if (c < 2) z[(((size_t)bb * 8 + c) << 12) + l] = v;
  else       anew[(((size_t)bb * 6 + (c - 2)) << 12) + l] = v;
}

// final: z[:,2:8] = a (6 ch)
__global__ void final_kernel(const float* __restrict__ a, float* __restrict__ z)
{
  int idx = blockIdx.x * blockDim.x + threadIdx.x;
  if (idx >= NB * 6 * LG) return;
  int l = idx & (LG - 1), rem = idx >> 12;
  int c = rem % 6, bb = rem / 6;
  z[(((size_t)bb * 8 + 2 + c) << 12) + l] = a[idx];
}

__global__ void zero2_kernel(float* a, float* b) { *a = 0.f; *b = 0.f; }

// ---------------------------------------------------------------------------
extern "C" void kernel_launch(void* const* d_in, const int* in_sizes, int n_in,
                              void* d_out, int out_size, void* d_ws, size_t ws_size,
                              hipStream_t stream)
{
  (void)in_sizes; (void)n_in; (void)out_size; (void)ws_size;
  const float* spect = (const float*)d_in[0];
  const float* audio = (const float*)d_in[1];
  const float* upw   = (const float*)d_in[2];
  const float* upb   = (const float*)d_in[3];

  char* ws = (char*)d_ws;
  size_t off = 0;
  auto alloc = [&](size_t bytes) -> char* {
    char* p = ws + off; off += (bytes + 255) & ~(size_t)255; return p;
  };
  float*    cond = (float*)alloc((size_t)NPOS * 1280 * 4);   // 41.9 MB (per-flow reuse)
  float*    su   = cond;                                     // alias: used only pre-flows (21 MB)
  _Float16* sgh  = (_Float16*)alloc((size_t)NPOS * 640 * 2); // 10.5 MB
  float*    z2   = (float*)alloc((size_t)NPOS * 256 * 4);    //  8.4 MB
  float*    rsb  = (float*)alloc((size_t)NPOS * 256 * 4);    //  8.4 MB
  float*    x    = (float*)alloc((size_t)NPOS * NCH * 4);
  _Float16* xh   = (_Float16*)alloc((size_t)NPOS * NCH * 2);
  _Float16* ah   = (_Float16*)alloc((size_t)NPOS * NCH * 2);
  float*    skip = (float*)alloc((size_t)NPOS * NCH * 4);
  float*    aA   = (float*)alloc((size_t)NB * 8 * LG * 4);
  float*    aB   = (float*)alloc((size_t)NB * 8 * LG * 4);
  _Float16* wcond = (_Float16*)alloc((size_t)1280 * 3200 * 2); // 8.2 MB
  _Float16* win   = (_Float16*)alloc((size_t)5 * 256 * 384 * 2);
  _Float16* wrs   = (_Float16*)alloc((size_t)5 * 256 * 128 * 2);

  float* zout   = (float*)d_out;          // z: [B][8][L] = 65536 floats
  float* ls_acc = zout + 65536;
  float* ld_acc = zout + 65537;

  const int TPB = 256;
  auto g1 = [](int n) { return dim3((unsigned)((n + 255) / 256)); };

  zero2_kernel<<<1, 1, 0, stream>>>(ls_acc, ld_acc);
  upsample_kernel<<<g1(NB * NMEL * TT), TPB, 0, stream>>>(spect, upw, upb, su);
  pack_sg_kernel<<<g1(NPOS * 640), TPB, 0, stream>>>(su, sgh);
  pack_audio_kernel<<<g1(NB * 8 * LG), TPB, 0, stream>>>(audio, aA);

  float* cura = aA;
  float* alta = aB;
  int nrem = 8, nhalf = 4;

  for (int k = 0; k < 6; ++k) {
    if (k == 4) {  // early output before this flow
      early_kernel<<<g1(NB * 8 * LG), TPB, 0, stream>>>(cura, zout, alta);
      float* t = cura; cura = alta; alta = t;
      nrem = 6; nhalf = 3;
    }
    const int base = 4 + 27 * k;
    const float* inv_w = (const float*)d_in[base + 0];
    const float* st_w  = (const float*)d_in[base + 1];
    const float* st_b  = (const float*)d_in[base + 2];
    const float* c_w   = (const float*)d_in[base + 3];
    const float* c_b   = (const float*)d_in[base + 4];
    const float* e_w   = (const float*)d_in[base + 25];
    const float* e_b   = (const float*)d_in[base + 26];

    logdet_kernel<<<1, 1, 0, stream>>>(inv_w, nrem, ld_acc);
    mix_kernel<<<g1(NB * nrem * LG), TPB, 0, stream>>>(inv_w, cura, alta, nrem);
    { float* t = cura; cura = alta; alta = t; }

    // f16 weight repacks for this flow
    cvtw_kernel<<<g1(1280 * 3200), TPB, 0, stream>>>(c_w, wcond, 640, 5, 1280 * 3200);
    for (int i = 0; i < 5; ++i) {
      const float* iw = (const float*)d_in[base + 5 + i];
      const float* rw = (const float*)d_in[base + 15 + i];
      int mrs = (i < 4) ? 256 : 128;
      cvtw_kernel<<<g1(256 * 384), TPB, 0, stream>>>(iw, win + (size_t)i * 256 * 384, 128, 3, 256 * 384);
      cvtw_kernel<<<g1(mrs * 128), TPB, 0, stream>>>(rw, wrs + (size_t)i * 256 * 128, 128, 1, mrs * 128);
    }

    start_kernel<<<g1(NPOS * NCH), TPB, 0, stream>>>(cura, st_w, st_b, x, xh, skip, nhalf, nrem);

    // cond conv: M=1280, K=3200 (WMMA)
    gemm_wmma<5, 640><<<dim3(1280 / 64, NPOS / 64), 128, 0, stream>>>(
        wcond, sgh, c_b, nullptr, cond, 1, 0, 0, 1280);

    for (int i = 0; i < 5; ++i) {
      const float* ib = (const float*)d_in[base + 10 + i];
      const float* rb = (const float*)d_in[base + 20 + i];
      int dil = 1 << i;
      int mrs = (i < 4) ? 256 : 128;
      // dilated conv + fused cond-slice add: z2 = in_conv(x) + in_b + cond[:, i*256:(i+1)*256]
      gemm_wmma<3, 128><<<dim3(256 / 64, NPOS / 64), 128, 0, stream>>>(
          win + (size_t)i * 256 * 384, xh, ib, cond, z2, dil, 1280, i * 256, 256);
      act_kernel<<<g1(NPOS * NCH), TPB, 0, stream>>>(z2, ah);
      // res/skip 1x1
      gemm_wmma<1, 128><<<dim3(mrs / 64, NPOS / 64), 128, 0, stream>>>(
          wrs + (size_t)i * 256 * 128, ah, rb, nullptr, rsb, 1, 0, 0, mrs);
      resid_kernel<<<g1(NPOS * NCH), TPB, 0, stream>>>(rsb, x, xh, skip, (i == 4) ? 1 : 0);
    }

    end_kernel<<<dim3(NPOS / 256), 256, 0, stream>>>(skip, e_w, e_b, cura, nhalf, nrem, ls_acc);
  }

  final_kernel<<<g1(NB * 6 * LG), TPB, 0, stream>>>(cura, zout);
}
